// LinearAttention_37890201485337
// MI455X (gfx1250) — matrix operand order repaired and checked by reading, not compiled, not run on hardware
//
#include <hip/hip_runtime.h>

// ---------------------------------------------------------------------------
// Problem dims (fixed by the reference)
// ---------------------------------------------------------------------------
#define BD 4
#define TD 4096
#define DD 1024
#define HH 16
#define DHD 64
#define LCH 64           // chunk length
#define CCH (TD / LCH)   // 64 chunks

typedef __attribute__((ext_vector_type(16))) __bf16 bf16x16;
typedef __attribute__((ext_vector_type(8)))  __bf16 bf16x8;
typedef __attribute__((ext_vector_type(8)))  float  floatx8;
typedef __attribute__((ext_vector_type(4)))  unsigned int uint32x4;
typedef __attribute__((ext_vector_type(8)))  int int32x8;
typedef __attribute__((ext_vector_type(4)))  int int32x4;

#if defined(__AMDGCN__) && __has_builtin(__builtin_amdgcn_global_load_async_to_lds_b128)
#define HAVE_ASYNC 1
#else
#define HAVE_ASYNC 0
#endif
#if defined(__AMDGCN__) && __has_builtin(__builtin_amdgcn_tensor_load_to_lds)
#define HAVE_TDM 1
#else
#define HAVE_TDM 0
#endif

__device__ __forceinline__ __bf16 to_bf(float f) { return (__bf16)f; }

// ---------------------------------------------------------------------------
// Async global->LDS copy of 16 bytes (falls back to direct copy).
// Builtin signature (from compiler diagnostic): param 0 is
// `int __vector(4) addrspace(1)*`, so cast to exactly that shape.
// ---------------------------------------------------------------------------
typedef int vec4i __attribute__((__vector_size__(16)));
typedef __attribute__((address_space(1))) vec4i gvec4i_t;
typedef __attribute__((address_space(3))) vec4i lvec4i_t;

__device__ __forceinline__ void copy16_async(const __bf16* g, __bf16* l) {
#if HAVE_ASYNC
  __builtin_amdgcn_global_load_async_to_lds_b128((gvec4i_t*)g, (lvec4i_t*)l,
                                                 0, 0);
#else
  *(bf16x8*)l = *(const bf16x8*)g;
#endif
}
__device__ __forceinline__ void wait_async0() {
#if HAVE_ASYNC
#if __has_builtin(__builtin_amdgcn_s_wait_asynccnt)
  __builtin_amdgcn_s_wait_asynccnt(0);
#else
  asm volatile("s_wait_asynccnt 0x0" ::: "memory");
#endif
#endif
}
__device__ __forceinline__ void wait_async4() {
#if HAVE_ASYNC
#if __has_builtin(__builtin_amdgcn_s_wait_asynccnt)
  __builtin_amdgcn_s_wait_asynccnt(4);
#else
  asm volatile("s_wait_asynccnt 0x4" ::: "memory");
#endif
#endif
}

// ---------------------------------------------------------------------------
// Tensor Data Mover: 2D bf16 tile (rows x cols), row stride in elements.
// D# group0: count=1 | lds_addr | global_addr | type=2
// D# group1: data_size=2B, tensor_dim0/1 = cols/rows, tile_dim0/1 = cols/rows,
//            tensor_dim0_stride = rstride
// ---------------------------------------------------------------------------
#if HAVE_TDM
__device__ __forceinline__ void tdm_load_tile_bf16(const __bf16* gsrc,
                                                   const __bf16* ldst,
                                                   int rows, int cols,
                                                   int rstride) {
  unsigned long long ga = (unsigned long long)(size_t)gsrc;
  unsigned lds = (unsigned)(size_t)ldst;  // low 32 bits of generic = LDS offset
  uint32x4 g0 = {1u, lds, (unsigned)ga,
                 (unsigned)((ga >> 32) & 0x01ffffffu) | (2u << 30)};
  int32x8 g1 = {(int)(1u << 16),            // data_size = 1 (2 bytes)
                (int)((cols & 0xffff) << 16),   // tensor_dim0 lo16 @ [63:48]
                (int)((rows & 0xffff) << 16),   // tensor_dim1 lo16 @ [95:80]
                (int)((cols & 0xffff) << 16),   // tile_dim0 @ [127:112]
                (rows & 0xffff),                // tile_dim1 @ [143:128]
                rstride,                        // tensor_dim0_stride lo32
                0, 0};
  int32x4 z4 = {0, 0, 0, 0};
#if __clang_major__ >= 23
  int32x8 z8 = {0, 0, 0, 0, 0, 0, 0, 0};
  __builtin_amdgcn_tensor_load_to_lds(g0, g1, z4, z4, z8, 0);
#else
  __builtin_amdgcn_tensor_load_to_lds(g0, g1, z4, z4, 0);
#endif
}
#endif

// ---------------------------------------------------------------------------
// WMMA helpers: v_wmma_f32_16x16x32_bf16, D = A(16x32) * B(32x16) + C
// ---------------------------------------------------------------------------
__device__ __forceinline__ floatx8 wmma_bf16(bf16x16 a, bf16x16 b, floatx8 c) {
  return __builtin_amdgcn_wmma_f32_16x16x32_bf16(
      false, a, false, b, (short)0, c, false, false);
}

__device__ __forceinline__ bf16x16 load_a_frag(const __bf16* tile, int stride) {
  int l  = threadIdx.x & 31;
  int m  = l & 15;
  int kb = (l >> 4) << 3;                    // 0 or 8
  const __bf16* p = tile + m * stride + kb;
  bf16x8 lo = *(const bf16x8*)p;             // K = kb .. kb+7
  bf16x8 hi = *(const bf16x8*)(p + 16);      // K = 16+kb .. 16+kb+7
  return __builtin_shufflevector(lo, hi, 0,1,2,3,4,5,6,7,8,9,10,11,12,13,14,15);
}

__device__ __forceinline__ bf16x16 load_b_frag(const __bf16* tile, int stride) {
  int l  = threadIdx.x & 31;
  int n  = l & 15;
  int kb = (l >> 4) << 3;                    // 0 or 8
  const __bf16* p = tile + n * stride + kb;
  bf16x8 lo = *(const bf16x8*)p;
  bf16x8 hi = *(const bf16x8*)(p + 16);
  return __builtin_shufflevector(lo, hi, 0,1,2,3,4,5,6,7,8,9,10,11,12,13,14,15);
}

// ---------------------------------------------------------------------------
// fp32 -> bf16 convert (grid-stride)
// ---------------------------------------------------------------------------
__global__ void convert_bf16_kernel(const float* __restrict__ in,
                                    __bf16* __restrict__ out, int n) {
  for (int i = blockIdx.x * blockDim.x + threadIdx.x; i < n;
       i += gridDim.x * blockDim.x)
    out[i] = to_bf(in[i]);
}

// ---------------------------------------------------------------------------
// 1024x1024 fp32 -> transposed bf16 (Wt[n][k] = W[k][n]); 32x32 LDS tiles
// ---------------------------------------------------------------------------
__global__ __launch_bounds__(1024) void transpose_bf16_kernel(
    const float* __restrict__ in, __bf16* __restrict__ out) {
  __shared__ __align__(16) __bf16 tile[32][33];
  int tx = threadIdx.x, ty = threadIdx.y;
  int x = blockIdx.x * 32 + tx;   // n
  int y = blockIdx.y * 32 + ty;   // k
  tile[ty][tx] = to_bf(in[y * DD + x]);
  __syncthreads();
  int ox = blockIdx.y * 32 + tx;  // k
  int oy = blockIdx.x * 32 + ty;  // n
  out[(size_t)oy * DD + ox] = tile[tx][ty];
}

// ---------------------------------------------------------------------------
// bf16 GEMM: C(MxN) = scale * A(MxK) * Bt(NxK)^T, fp32 accumulate via WMMA.
// Block tile 128x128, BK=32, double-buffered LDS with async global->LDS.
// 256 threads = 8 waves, each wave owns a 32x64 sub-tile (2x4 WMMA tiles).
// ---------------------------------------------------------------------------
template <bool OUT_F32>
__global__ __launch_bounds__(256) void gemm_bf16_kernel(
    const __bf16* __restrict__ A, const __bf16* __restrict__ Bt,
    void* __restrict__ Cout, int M, int N, int K, float scale) {
  __shared__ __align__(16) __bf16 As[2][128 * 32];
  __shared__ __align__(16) __bf16 Bs[2][128 * 32];

  const int rowBase = blockIdx.x * 128;
  const int colBase = blockIdx.y * 128;
  const int t  = threadIdx.x;
  const int w  = t >> 5;
  const int wr = w & 3;        // 0..3 -> 32-row band
  const int wc = w >> 2;       // 0..1 -> 64-col band
  const int l  = t & 31;
  const int nloc = l & 15;
  const int hi8  = (l >> 4) * 8;

  floatx8 acc[2][4];
#pragma unroll
  for (int i = 0; i < 2; ++i)
#pragma unroll
    for (int j = 0; j < 4; ++j) acc[i][j] = {};

  const int lr = t >> 1;             // 0..127
  const int lc = (t & 1) * 16;       // 0 or 16

  auto issue_tile = [&](int buf, int k0) {
    const __bf16* srcA = A + (size_t)(rowBase + lr) * K + k0 + lc;
    copy16_async(srcA,     &As[buf][lr * 32 + lc]);
    copy16_async(srcA + 8, &As[buf][lr * 32 + lc + 8]);
    const __bf16* srcB = Bt + (size_t)(colBase + lr) * K + k0 + lc;
    copy16_async(srcB,     &Bs[buf][lr * 32 + lc]);
    copy16_async(srcB + 8, &Bs[buf][lr * 32 + lc + 8]);
    if (k0 + 32 < K)  // hint the tile after next toward L2
      __builtin_prefetch(srcA + 32, 0, 1);
  };

  issue_tile(0, 0);
  for (int k0 = 0; k0 < K; k0 += 32) {
    const int buf = (k0 >> 5) & 1;
    const bool have_next = (k0 + 32) < K;
    if (have_next) {
      issue_tile(buf ^ 1, k0 + 32);
      wait_async4();   // previous tile's 4 async ops complete (in-order)
    } else {
      wait_async0();
    }
    __syncthreads();

    bf16x16 af[2], bfm[4];
#pragma unroll
    for (int mt = 0; mt < 2; ++mt)
      af[mt] = load_a_frag(&As[buf][(wr * 32 + mt * 16) * 32], 32);
#pragma unroll
    for (int nt = 0; nt < 4; ++nt)
      bfm[nt] = load_b_frag(&Bs[buf][(wc * 64 + nt * 16) * 32], 32);
#pragma unroll
    for (int mt = 0; mt < 2; ++mt)
#pragma unroll
      for (int nt = 0; nt < 4; ++nt)
        acc[mt][nt] = wmma_bf16(af[mt], bfm[nt], acc[mt][nt]);
    __syncthreads();   // safe to overwrite buf next iteration
  }

#pragma unroll
  for (int mt = 0; mt < 2; ++mt)
#pragma unroll
    for (int nt = 0; nt < 4; ++nt)
#pragma unroll
      for (int rr = 0; rr < 8; ++rr) {
        int row = rowBase + wr * 32 + mt * 16 + rr + hi8;
        int col = colBase + wc * 64 + nt * 16 + nloc;
        float v = acc[mt][nt][rr] * scale;
        if (OUT_F32)
          ((float*)Cout)[(size_t)row * N + col] = v;
        else
          ((__bf16*)Cout)[(size_t)row * N + col] = to_bf(v);
      }
}

// ---------------------------------------------------------------------------
// Chunked retention core. One block per (b,h); 64 chunks sequential.
// Per chunk (all 64x64):
//   A      = q . k^T  -> causal decay mask -> Amk (bf16)
//   o      = rowscale(gamma^(i+1)) * (q . S_prev) + Amk . v
//   S_new  = gamma^L * S + (k * gamma^(L-1-j))^T . v      (fp32 state)
// Decay powers come from an LDS table gamma^p, p in [0,71].
// ---------------------------------------------------------------------------
__global__ __launch_bounds__(256) void retention_kernel(
    const __bf16* __restrict__ q, const __bf16* __restrict__ k,
    const __bf16* __restrict__ v, __bf16* __restrict__ o) {
  __shared__ __align__(16) __bf16 qs [64 * 64];  // q[i][d]
  __shared__ __align__(16) __bf16 ks [64 * 64];  // k[j][d]
  __shared__ __align__(16) __bf16 ktd[64 * 64];  // decayed k^T [d][j]
  __shared__ __align__(16) __bf16 vts[64 * 64];  // v^T [e][j]
  __shared__ __align__(16) __bf16 Amk[64 * 64];  // masked A [i][j]
  __shared__ __align__(16) __bf16 Sb [64 * 64];  // bf16 S^T [e][d]
  __shared__ __align__(16) float  Sf [64 * 64];  // fp32 S^T [e][d]
  __shared__ float powtab[72];                   // gamma^p

  const int bh = blockIdx.x;
  const int b = bh >> 4, h = bh & 15;
  const float gamma = 1.0f - exp2f(-5.0f - (float)h);
  const float lg = log2f(gamma);
  const float gL = exp2f(64.0f * lg);

  const int t = threadIdx.x;
  if (t < 72) powtab[t] = exp2f((float)t * lg);
  for (int i = t; i < 64 * 64; i += 256) { Sf[i] = 0.0f; Sb[i] = to_bf(0.0f); }
  __syncthreads();

  const int w = t >> 5, l = t & 31;
  const int tid0 = w * 2;
  const int nloc = l & 15;
  const int hi8  = (l >> 4) * 8;

  const size_t base = (size_t)b * TD * DD + (size_t)h * DHD;

  const int lrow = t >> 2;           // 0..63
  const int lcol = (t & 3) * 16;     // 0,16,32,48

  for (int c = 0; c < CCH; ++c) {
    const size_t gtile = base + (size_t)(c * 64) * DD;   // chunk tile origin
    const size_t g = gtile + (size_t)lrow * DD + lcol;

    // ---- load q/k (TDM when available) and scatter v^T ----------------
#if HAVE_TDM
    if (w == 0) {
      tdm_load_tile_bf16(q + gtile, qs, 64, 64, DD);
      tdm_load_tile_bf16(k + gtile, ks, 64, 64, DD);
      __builtin_amdgcn_s_wait_tensorcnt(0);
    }
#else
    *(bf16x8*)(qs + lrow * 64 + lcol)     = *(const bf16x8*)(q + g);
    *(bf16x8*)(qs + lrow * 64 + lcol + 8) = *(const bf16x8*)(q + g + 8);
    *(bf16x8*)(ks + lrow * 64 + lcol)     = *(const bf16x8*)(k + g);
    *(bf16x8*)(ks + lrow * 64 + lcol + 8) = *(const bf16x8*)(k + g + 8);
#endif
    bf16x8 v0 = *(const bf16x8*)(v + g), v1 = *(const bf16x8*)(v + g + 8);
#pragma unroll
    for (int u = 0; u < 8; ++u) {
      vts[(lcol + u)     * 64 + lrow] = v0[u];
      vts[(lcol + 8 + u) * 64 + lrow] = v1[u];
    }
    __syncthreads();

    // ---- phase A: A = q.k^T (mask+decay), o_pre = rowscale(q.S_prev) --
    floatx8 accO[2];
#pragma unroll
    for (int tt = 0; tt < 2; ++tt) {
      const int tid = tid0 + tt;
      const int mt = tid >> 2, nt = tid & 3;
      floatx8 aA = {};
      floatx8 aO = {};
#pragma unroll
      for (int kk = 0; kk < 2; ++kk) {
        bf16x16 af = load_a_frag(qs + (mt * 16) * 64 + kk * 32, 64);
        bf16x16 bk = load_b_frag(ks + (nt * 16) * 64 + kk * 32, 64);
        aA = wmma_bf16(af, bk, aA);
        bf16x16 bs = load_b_frag(Sb + (nt * 16) * 64 + kk * 32, 64);
        aO = wmma_bf16(af, bs, aO);
      }
#pragma unroll
      for (int rr = 0; rr < 8; ++rr) {
        const int i = mt * 16 + rr + hi8;
        const int j = nt * 16 + nloc;
        const int rel = i - j;
        const float dv = (rel >= 0) ? powtab[rel] : 0.0f;
        Amk[i * 64 + j] = to_bf(aA[rr] * dv);
        aO[rr] *= powtab[i + 1];                 // gamma^(i+1)
      }
      accO[tt] = aO;
    }
    // build decayed k^T from the LDS copy of k (read-read with phase A)
    {
      bf16x8 kr0 = *(const bf16x8*)(ks + lrow * 64 + lcol);
      bf16x8 kr1 = *(const bf16x8*)(ks + lrow * 64 + lcol + 8);
      const float kd = powtab[63 - lrow];        // gamma^(L-1-j)
#pragma unroll
      for (int u = 0; u < 8; ++u) {
        ktd[(lcol + u)     * 64 + lrow] = to_bf((float)kr0[u] * kd);
        ktd[(lcol + 8 + u) * 64 + lrow] = to_bf((float)kr1[u] * kd);
      }
    }
    __syncthreads();

    // ---- phase B: o += A.v ; kv = ktd.v ; S update --------------------
#pragma unroll
    for (int tt = 0; tt < 2; ++tt) {
      const int tid = tid0 + tt;
      const int mt = tid >> 2, nt = tid & 3;
      floatx8 aO = accO[tt];
      floatx8 kv = {};
#pragma unroll
      for (int kk = 0; kk < 2; ++kk) {
        bf16x16 bv  = load_b_frag(vts + (nt * 16) * 64 + kk * 32, 64);
        bf16x16 aa  = load_a_frag(Amk + (mt * 16) * 64 + kk * 32, 64);
        aO = wmma_bf16(aa, bv, aO);
        bf16x16 akt = load_a_frag(ktd + (mt * 16) * 64 + kk * 32, 64);
        kv = wmma_bf16(akt, bv, kv);
      }
#pragma unroll
      for (int rr = 0; rr < 8; ++rr) {
        const int i = mt * 16 + rr + hi8;   // output row / state d-index
        const int e = nt * 16 + nloc;       // output col / state e-index
        o[base + (size_t)(c * 64 + i) * DD + e] = to_bf(aO[rr]);
        float s = Sf[e * 64 + i] * gL + kv[rr];
        Sf[e * 64 + i] = s;
        Sb[e * 64 + i] = to_bf(s);
      }
    }
    __syncthreads();
  }
}

// ---------------------------------------------------------------------------
// Launcher
// ---------------------------------------------------------------------------
extern "C" void kernel_launch(void* const* d_in, const int* in_sizes, int n_in,
                              void* d_out, int out_size, void* d_ws,
                              size_t ws_size, hipStream_t stream) {
  (void)in_sizes; (void)n_in; (void)out_size; (void)ws_size;
  const float* query = (const float*)d_in[0];
  const float* Wq = (const float*)d_in[1];
  const float* Wk = (const float*)d_in[2];
  const float* Wv = (const float*)d_in[3];
  const float* Wo = (const float*)d_in[4];

  const size_t nX = (size_t)BD * TD * DD;      // 16,777,216 elements
  const size_t nW = (size_t)DD * DD;

  char* ws = (char*)d_ws;
  __bf16* Xbf = (__bf16*)ws; ws += nX * 2;
  __bf16* Wqt = (__bf16*)ws; ws += nW * 2;
  __bf16* Wkt = (__bf16*)ws; ws += nW * 2;
  __bf16* Wvt = (__bf16*)ws; ws += nW * 2;
  __bf16* Wot = (__bf16*)ws; ws += nW * 2;
  __bf16* qb  = (__bf16*)ws; ws += nX * 2;
  __bf16* kb  = (__bf16*)ws; ws += nX * 2;
  __bf16* vb  = (__bf16*)ws; ws += nX * 2;
  __bf16* ob  = (__bf16*)ws; ws += nX * 2;

  // 1) precision conversion + weight transposes
  convert_bf16_kernel<<<4096, 256, 0, stream>>>(query, Xbf, (int)nX);
  dim3 tb(32, 32), tg(32, 32);
  transpose_bf16_kernel<<<tg, tb, 0, stream>>>(Wq, Wqt);
  transpose_bf16_kernel<<<tg, tb, 0, stream>>>(Wk, Wkt);
  transpose_bf16_kernel<<<tg, tb, 0, stream>>>(Wv, Wvt);
  transpose_bf16_kernel<<<tg, tb, 0, stream>>>(Wo, Wot);

  // 2) q/k/v projections (q scaled by DH^-0.5 = 0.125)
  dim3 gg(128, 8);  // 16384/128 x 1024/128
  gemm_bf16_kernel<false><<<gg, 256, 0, stream>>>(Xbf, Wqt, qb,
      BD * TD, DD, DD, 0.125f);
  gemm_bf16_kernel<false><<<gg, 256, 0, stream>>>(Xbf, Wkt, kb,
      BD * TD, DD, DD, 1.0f);
  gemm_bf16_kernel<false><<<gg, 256, 0, stream>>>(Xbf, Wvt, vb,
      BD * TD, DD, DD, 1.0f);

  // 3) chunked retention, one block per (b,h)
  retention_kernel<<<BD * HH, 256, 0, stream>>>(qb, kb, vb, ob);

  // 4) output projection -> fp32 d_out
  gemm_bf16_kernel<true><<<gg, 256, 0, stream>>>(ob, Wot, d_out,
      BD * TD, DD, DD, 1.0f);
}
